// MeteorNet_14482629722290
// MI455X (gfx1250) — matrix operand order repaired and checked by reading, not compiled, not run on hardware
//
#include <hip/hip_runtime.h>
#include <hip/hip_bf16.h>
#include <math.h>

// ---------------------------------------------------------------------------
// MeteorNet on gfx1250 (MI455X). All 1x1-conv / linear layers run as f16
// WMMA GEMMs (v_wmma_f32_16x16x32_f16) with f32 accumulation; BN uses batch
// statistics computed on the fly; FPS / kNN / grouping / scatter-max /
// 3-NN interpolation are custom wave32 kernels.
//
// Tensor layout convention in workspace: channel-major  T[c][b*P + p].
// ---------------------------------------------------------------------------

typedef __attribute__((ext_vector_type(16))) _Float16 v16h;
typedef __attribute__((ext_vector_type(8)))  _Float16 v8h;
typedef __attribute__((ext_vector_type(8)))  float    v8f;

#define K_NN 32
#define BN_EPS 1e-3f

static inline int rup(int x, int m) { return ((x + m - 1) / m) * m; }

// --------------------------- ingest ---------------------------------------
// xyz [B][3][N] -> coords0 [3][B*N]; feat/times [B][1][N] -> l0p [2][B*N], times0
__global__ void ingest_kernel(const float* __restrict__ xyz,
                              const float* __restrict__ feat,
                              const float* __restrict__ times,
                              float* __restrict__ coords0,
                              float* __restrict__ times0,
                              float* __restrict__ l0p, int B, int N) {
  size_t i = (size_t)blockIdx.x * blockDim.x + threadIdx.x;
  size_t BN = (size_t)B * N;
  if (i >= BN) return;
  int b = (int)(i / N), p = (int)(i % N);
  for (int c = 0; c < 3; ++c)
    coords0[(size_t)c * BN + i] = xyz[((size_t)b * 3 + c) * N + p];
  float tv = times[(size_t)b * N + p];
  float fv = feat[(size_t)b * N + p];
  times0[i] = tv;
  l0p[i] = fv;
  l0p[BN + i] = tv;
}

// --------------------------- FPS ------------------------------------------
// one workgroup (512 threads) per batch; iterative farthest point sampling.
__global__ void fps_kernel(const float* __restrict__ xyz /*[3][B*N]*/,
                           int B, int N, int npoint,
                           int* __restrict__ outIdx /*[B][npoint]*/,
                           float* __restrict__ distWs /*[B*N]*/) {
  __shared__ float sdist[512];
  __shared__ int   sidx[512];
  int b = blockIdx.x;
  int tid = threadIdx.x;
  size_t BN = (size_t)B * N;
  const float* xs = xyz + (size_t)0 * BN + (size_t)b * N;
  const float* ys = xyz + (size_t)1 * BN + (size_t)b * N;
  const float* zs = xyz + (size_t)2 * BN + (size_t)b * N;
  float* dist = distWs + (size_t)b * N;
  for (int p = tid; p < N; p += 512) dist[p] = 3.4e38f;
  __syncthreads();
  int far = 0;
  for (int it = 0; it < npoint; ++it) {
    if (tid == 0) outIdx[(size_t)b * npoint + it] = far;
    float cx = xs[far], cy = ys[far], cz = zs[far];
    float best = -1.0f; int bestI = 0;
    for (int p = tid; p < N; p += 512) {
      float dx = xs[p] - cx, dy = ys[p] - cy, dz = zs[p] - cz;
      float d = dx * dx + dy * dy + dz * dz;
      float dm = fminf(dist[p], d);
      dist[p] = dm;
      if (dm > best) { best = dm; bestI = p; }
    }
    sdist[tid] = best; sidx[tid] = bestI;
    __syncthreads();
    for (int s = 256; s > 0; s >>= 1) {
      if (tid < s) {
        if (sdist[tid + s] > sdist[tid] ||
            (sdist[tid + s] == sdist[tid] && sidx[tid + s] < sidx[tid])) {
          sdist[tid] = sdist[tid + s]; sidx[tid] = sidx[tid + s];
        }
      }
      __syncthreads();
    }
    far = sidx[0];
    __syncthreads();
  }
}

// --------------------------- gather ---------------------------------------
// dst[c][b*n + j] = src[c][b*N + idx[b*n + j]]
__global__ void gather_rows(const float* __restrict__ src, const int* __restrict__ idx,
                            float* __restrict__ dst, int C, int B, int N, int n) {
  size_t i = (size_t)blockIdx.x * blockDim.x + threadIdx.x;
  size_t tot = (size_t)C * B * n;
  if (i >= tot) return;
  int col = (int)(i % ((size_t)B * n));
  int ch  = (int)(i / ((size_t)B * n));
  int b = col / n;
  dst[i] = src[(size_t)ch * B * N + (size_t)b * N + idx[col]];
}

// --------------------------- kNN ------------------------------------------
// one wave32 per (b, centroid); 32 rounds of lexicographic (d, idx) wave-min
// with exclusion of already-selected points (matches top_k stability).
__global__ void knn_kernel(const float* __restrict__ centers /*[3][B*n]*/,
                           const float* __restrict__ src /*[3][B*N]*/,
                           int B, int n, int N, int* __restrict__ nind /*[B*n*K]*/) {
  int gwave = (int)(((size_t)blockIdx.x * blockDim.x + threadIdx.x) >> 5);
  int lane = threadIdx.x & 31;
  if (gwave >= B * n) return;
  int b = gwave / n, i = gwave % n;
  size_t Bn = (size_t)B * n, BN = (size_t)B * N;
  float cx = centers[0 * Bn + (size_t)b * n + i];
  float cy = centers[1 * Bn + (size_t)b * n + i];
  float cz = centers[2 * Bn + (size_t)b * n + i];
  const float* sx = src + 0 * BN + (size_t)b * N;
  const float* sy = src + 1 * BN + (size_t)b * N;
  const float* sz = src + 2 * BN + (size_t)b * N;
  float lastD = -1.0f; int lastP = -1;
  for (int r = 0; r < K_NN; ++r) {
    float bd = 3.4e38f; int bp = 0x7fffffff;
    for (int p = lane; p < N; p += 32) {
      float dx = sx[p] - cx, dy = sy[p] - cy, dz = sz[p] - cz;
      float d = dx * dx + dy * dy + dz * dz;
      bool excluded = (d < lastD) || (d == lastD && p <= lastP);
      if (!excluded && (d < bd || (d == bd && p < bp))) { bd = d; bp = p; }
    }
    for (int off = 16; off > 0; off >>= 1) {
      float od = __shfl_xor(bd, off, 32);
      int   op = __shfl_xor(bp, off, 32);
      if (od < bd || (od == bd && op < bp)) { bd = od; bp = op; }
    }
    if (lane == 0) nind[((size_t)b * n + i) * K_NN + r] = bp;
    lastD = bd; lastP = bp;
  }
}

// --------------------------- grouping --------------------------------------
// grouped [3+1+Cf][B*n*K] = concat(xg - center, tg, fg)
__global__ void group_kernel(const float* __restrict__ srcXyz /*[3][B*N]*/,
                             const float* __restrict__ srcT /*[B*N]*/,
                             const float* __restrict__ srcF /*[Cf][B*N]*/,
                             const float* __restrict__ centers /*[3][B*n]*/,
                             const int* __restrict__ nind,
                             float* __restrict__ grouped,
                             int B, int N, int n, int Cf) {
  size_t col = (size_t)blockIdx.x * blockDim.x + threadIdx.x;
  size_t BnK = (size_t)B * n * K_NN;
  if (col >= BnK) return;
  int bi = (int)(col / K_NN);
  int i = bi % n, b = bi / n;
  int p = nind[col];
  size_t BN = (size_t)B * N, Bn = (size_t)B * n;
  for (int c = 0; c < 3; ++c)
    grouped[(size_t)c * BnK + col] =
        srcXyz[(size_t)c * BN + (size_t)b * N + p] - centers[(size_t)c * Bn + (size_t)b * n + i];
  grouped[3 * BnK + col] = srcT[(size_t)b * N + p];
  for (int c = 0; c < Cf; ++c)
    grouped[(size_t)(4 + c) * BnK + col] = srcF[(size_t)c * BN + (size_t)b * N + p];
}

// --------------------------- f16 packing -----------------------------------
// Wh [Mpad][Kpad] (zero padded) from W [M][K] row-major.
__global__ void pack_w(const float* __restrict__ W, _Float16* __restrict__ Wh,
                       int M, int K, int Mpad, int Kpad) {
  size_t i = (size_t)blockIdx.x * blockDim.x + threadIdx.x;
  size_t tot = (size_t)Mpad * Kpad;
  if (i >= tot) return;
  int m = (int)(i / Kpad), cc = (int)(i % Kpad);
  Wh[i] = (m < M && cc < K) ? (_Float16)W[(size_t)m * K + cc] : (_Float16)0.0f;
}
// XhT [Ppad][Cpad] (transposed, zero padded) from X [C][P] channel-major.
__global__ void pack_x(const float* __restrict__ X, _Float16* __restrict__ XhT,
                       int C, int P, int Npad, int Cpad) {
  size_t i = (size_t)blockIdx.x * blockDim.x + threadIdx.x;
  size_t tot = (size_t)Npad * Cpad;
  if (i >= tot) return;
  int p = (int)(i / Cpad), cc = (int)(i % Cpad);
  XhT[i] = (cc < C && p < P) ? (_Float16)X[(size_t)cc * P + p] : (_Float16)0.0f;
}

// --------------------------- WMMA GEMM -------------------------------------
// Y[M][P] = Wh[Mpad][Kpad] * XhT[Npad][Kpad]^T + bias.
// Each wave computes a 16(M) x 64(N) strip: one A fragment is reused across
// 4 B fragments / 4 f32 accumulators per K-step (register blocking -> ~2x
// arithmetic intensity vs one-tile-per-wave).
// A-fragment (16x32 f16): lanes 0-15 hold M=lane, K {0..7,16..23}; lanes 16-31
//   hold K {8..15,24..31}  -> two contiguous 16B loads from the Wh row.
// B-fragment (32x16 f16): lanes 0-15 hold N=lane, K {0..7,16..23}; lanes 16-31
//   hold K {8..15,24..31} -> two 16B loads from the XhT row (thanks to the transpose).
__global__ void wmma_gemm(const _Float16* __restrict__ Wh,
                          const _Float16* __restrict__ XhT,
                          const float* __restrict__ bias,
                          float* __restrict__ Y,
                          int M, int P, int Kpad, int Npad, int Mpad) {
  int wave = (int)(((size_t)blockIdx.x * blockDim.x + threadIdx.x) >> 5);
  int lane = threadIdx.x & 31;
  int tilesM = Mpad >> 4, stripsN = Npad >> 6;  // Npad is a multiple of 64
  int total = tilesM * stripsN;
  int nwaves = (int)(((size_t)gridDim.x * blockDim.x) >> 5);
  int half = lane >> 4;
  for (int t = wave; t < total; t += nwaves) {
    int tm = t % tilesM, tn = t / tilesM;
    int mBase = tm << 4, nBase = tn << 6;
    const _Float16* aPtr = Wh + (size_t)(mBase + (lane & 15)) * Kpad + half * 8;
    const _Float16* bPtr0 = XhT + (size_t)(nBase + (lane & 15)) * Kpad + half * 8;
    v8f acc0 = {}, acc1 = {}, acc2 = {}, acc3 = {};
    size_t bStride = (size_t)16 * Kpad;
    for (int k0 = 0; k0 < Kpad; k0 += 32) {
      __builtin_prefetch(aPtr + k0 + 128, 0, 1);
      v8h a0 = *(const v8h*)(aPtr + k0);
      v8h a1 = *(const v8h*)(aPtr + k0 + 16);
      v16h a;
#pragma unroll
      for (int q = 0; q < 8; ++q) { a[q] = a0[q]; a[8 + q] = a1[q]; }
#pragma unroll
      for (int j = 0; j < 4; ++j) {
        const _Float16* bp = bPtr0 + (size_t)j * bStride + k0;
        __builtin_prefetch(bp + 128, 0, 1);
        v8h b0 = *(const v8h*)(bp);
        v8h b1 = *(const v8h*)(bp + 16);
        v16h bfrag;
#pragma unroll
        for (int q = 0; q < 8; ++q) { bfrag[q] = b0[q]; bfrag[8 + q] = b1[q]; }
        v8f cin = (j == 0) ? acc0 : (j == 1) ? acc1 : (j == 2) ? acc2 : acc3;
        v8f d = __builtin_amdgcn_wmma_f32_16x16x32_f16(
            /*neg_a=*/false, a, /*neg_b=*/false, bfrag,
            /*c_mod=*/(short)0, cin, /*reuse_a=*/false, /*reuse_b=*/false);
        if (j == 0) acc0 = d; else if (j == 1) acc1 = d; else if (j == 2) acc2 = d; else acc3 = d;
      }
    }
    int mLoc = half << 3;
#pragma unroll
    for (int j = 0; j < 4; ++j) {
      int col = nBase + (j << 4) + (lane & 15);
      if (col >= P) continue;
      v8f acc = (j == 0) ? acc0 : (j == 1) ? acc1 : (j == 2) ? acc2 : acc3;
#pragma unroll
      for (int r = 0; r < 8; ++r) {
        int m = mBase + mLoc + r;
        if (m < M) Y[(size_t)m * P + col] = acc[r] + bias[m];
      }
    }
  }
}

// --------------------------- BatchNorm + activation -------------------------
__global__ void bn_stats(const float* __restrict__ Y, float* __restrict__ mean,
                         float* __restrict__ istd, int P) {
  __shared__ float ssum[256], ssq[256];
  int c = blockIdx.x, tid = threadIdx.x;
  const float* row = Y + (size_t)c * P;
  float s = 0.0f, q = 0.0f;
  for (int p = tid; p < P; p += 256) { float v = row[p]; s += v; q += v * v; }
  ssum[tid] = s; ssq[tid] = q;
  __syncthreads();
  for (int st = 128; st > 0; st >>= 1) {
    if (tid < st) { ssum[tid] += ssum[tid + st]; ssq[tid] += ssq[tid + st]; }
    __syncthreads();
  }
  if (tid == 0) {
    float m = ssum[0] / P;
    float v = ssq[0] / P - m * m;
    if (v < 0.0f) v = 0.0f;
    mean[c] = m;
    istd[c] = rsqrtf(v + BN_EPS);
  }
}

__global__ void bn_act(float* __restrict__ Y, const float* __restrict__ mean,
                       const float* __restrict__ istd, const float* __restrict__ g,
                       const float* __restrict__ bt, int P, size_t total, float leak) {
  size_t i = (size_t)blockIdx.x * blockDim.x + threadIdx.x;
  if (i >= total) return;
  int ch = (int)(i / P);
  float x = (Y[i] - mean[ch]) * istd[ch] * g[ch] + bt[ch];
  Y[i] = x > 0.0f ? x : x * leak;
}

// --------------------------- temporal scatter-max ---------------------------
// u[t*Cc+ch][b*n+i] = max(0, max_{j: time(nind)==t} nf[ch][(b*n+i)*K + j])
__global__ void scatter_max(const float* __restrict__ nf /*[Cc][B*n*K]*/,
                            const float* __restrict__ srcT /*[B*N]*/,
                            const int* __restrict__ nind,
                            float* __restrict__ u /*[2Cc][B*n]*/,
                            int B, int N, int n, int Cc) {
  size_t i = (size_t)blockIdx.x * blockDim.x + threadIdx.x;
  size_t Bn = (size_t)B * n;
  size_t tot = (size_t)Cc * Bn;
  if (i >= tot) return;
  int col = (int)(i % Bn);
  int ch = (int)(i / Bn);
  int b = col / n;
  size_t BnK = Bn * K_NN;
  float m0 = -3.4e38f, m1 = -3.4e38f;
  for (int j = 0; j < K_NN; ++j) {
    int p = nind[(size_t)col * K_NN + j];
    float tv = srcT[(size_t)b * N + p];
    float v = nf[(size_t)ch * BnK + (size_t)col * K_NN + j];
    if (tv < 0.5f) { if (v > m0) m0 = v; } else { if (v > m1) m1 = v; }
  }
  u[(size_t)ch * Bn + col] = fmaxf(m0, 0.0f);
  u[((size_t)Cc + ch) * Bn + col] = fmaxf(m1, 0.0f);
}

// --------------------------- FP interpolation -------------------------------
// out[0..C1-1] = inverse-sq-dist 3-NN interp of feat1; out[C1..] = feat2.
__global__ void fp_interp(const float* __restrict__ xyz2 /*[3][B*N2]*/,
                          const float* __restrict__ xyz1 /*[3][B*N1]*/,
                          const float* __restrict__ feat1 /*[C1][B*N1]*/,
                          const float* __restrict__ feat2 /*[C2][B*N2]*/,
                          float* __restrict__ out /*[C1+C2][B*N2]*/,
                          int B, int N2, int N1, int C1, int C2) {
  size_t col = (size_t)blockIdx.x * blockDim.x + threadIdx.x;
  size_t BN2 = (size_t)B * N2, BN1 = (size_t)B * N1;
  if (col >= BN2) return;
  int b = (int)(col / N2);
  float cx = xyz2[0 * BN2 + col], cy = xyz2[1 * BN2 + col], cz = xyz2[2 * BN2 + col];
  const float* sx = xyz1 + 0 * BN1 + (size_t)b * N1;
  const float* sy = xyz1 + 1 * BN1 + (size_t)b * N1;
  const float* sz = xyz1 + 2 * BN1 + (size_t)b * N1;
  float d0 = 3.4e38f, d1 = 3.4e38f, d2 = 3.4e38f;
  int i0 = 0, i1 = 0, i2 = 0;
  for (int p = 0; p < N1; ++p) {
    float dx = sx[p] - cx, dy = sy[p] - cy, dz = sz[p] - cz;
    float d = dx * dx + dy * dy + dz * dz;
    if (d < d0) { d2 = d1; i2 = i1; d1 = d0; i1 = i0; d0 = d; i0 = p; }
    else if (d < d1) { d2 = d1; i2 = i1; d1 = d; i1 = p; }
    else if (d < d2) { d2 = d; i2 = p; }
  }
  float w0 = 1.0f / fmaxf(d0, 1e-10f);
  float w1 = 1.0f / fmaxf(d1, 1e-10f);
  float w2 = 1.0f / fmaxf(d2, 1e-10f);
  float ws = w0 + w1 + w2;
  w0 /= ws; w1 /= ws; w2 /= ws;
  for (int c = 0; c < C1; ++c) {
    const float* row = feat1 + (size_t)c * BN1 + (size_t)b * N1;
    out[(size_t)c * BN2 + col] = w0 * row[i0] + w1 * row[i1] + w2 * row[i2];
  }
  for (int c = 0; c < C2; ++c)
    out[(size_t)(C1 + c) * BN2 + col] = feat2[(size_t)c * BN2 + col];
}

// --------------------------- output slice -----------------------------------
// pred [C][B*N] -> out[b][o][0:N/2]
__global__ void slice_out(const float* __restrict__ Ypred, float* __restrict__ out,
                          int B, int N, int C) {
  int half = N / 2;
  size_t i = (size_t)blockIdx.x * blockDim.x + threadIdx.x;
  size_t tot = (size_t)B * C * half;
  if (i >= tot) return;
  int p = (int)(i % half);
  int o = (int)((i / half) % C);
  int b = (int)(i / ((size_t)half * C));
  out[i] = Ypred[(size_t)o * B * N + (size_t)b * N + p];
}

// ===========================================================================
// Host orchestration
// ===========================================================================
struct ConvLayer { const float *w, *b, *g, *bt; };

struct Ctx {
  hipStream_t st;
  int B;
  float *bufA, *bufB, *mean, *istd, *fpsDist;
  _Float16 *XhT, *Wh;
  int *fpsIdx, *nind;
};

static void launch_layer(Ctx& c, const ConvLayer& L, const float* X, float* Y,
                         int Cin, int Cout, int P, bool bn, float leak) {
  int Kpad = rup(Cin, 32), Npad = rup(P, 64), Mpad = rup(Cout, 16);
  {
    size_t n = (size_t)Mpad * Kpad;
    pack_w<<<(unsigned)((n + 255) / 256), 256, 0, c.st>>>(L.w, c.Wh, Cout, Cin, Mpad, Kpad);
  }
  {
    size_t n = (size_t)Npad * Kpad;
    pack_x<<<(unsigned)((n + 255) / 256), 256, 0, c.st>>>(X, c.XhT, Cin, P, Npad, Kpad);
  }
  {
    int tiles = (Mpad / 16) * (Npad / 64);
    wmma_gemm<<<(unsigned)((tiles + 3) / 4), 128, 0, c.st>>>(c.Wh, c.XhT, L.b, Y,
                                                             Cout, P, Kpad, Npad, Mpad);
  }
  if (bn) {
    bn_stats<<<(unsigned)Cout, 256, 0, c.st>>>(Y, c.mean, c.istd, P);
    size_t n = (size_t)Cout * P;
    bn_act<<<(unsigned)((n + 255) / 256), 256, 0, c.st>>>(Y, c.mean, c.istd, L.g, L.bt,
                                                          P, n, leak);
  }
}

// runs an nL-layer MLP chain starting with X in bufA; returns buffer with result
static float* run_chain(Ctx& c, const ConvLayer* Ls, const int* ch, int nL, int P) {
  float* X = c.bufA;
  float* Y = c.bufB;
  for (int l = 0; l < nL; ++l) {
    launch_layer(c, Ls[l], X, Y, ch[l], ch[l + 1], P, true, 0.0f);
    float* t = X; X = Y; Y = t;
  }
  return X;
}

static void run_meteor(Ctx& c, const float* srcXyz, const float* srcT,
                       const float* srcF, int Cf, int Nsrc, int n,
                       const ConvLayer conv[3], const int convC[4],
                       const ConvLayer& unary, int cuOut,
                       float* xOut, float* tOut, float* pOut) {
  int B = c.B;
  fps_kernel<<<(unsigned)B, 512, 0, c.st>>>(srcXyz, B, Nsrc, n, c.fpsIdx, c.fpsDist);
  {
    size_t tot = (size_t)3 * B * n;
    gather_rows<<<(unsigned)((tot + 255) / 256), 256, 0, c.st>>>(srcXyz, c.fpsIdx, xOut, 3, B, Nsrc, n);
  }
  {
    size_t tot = (size_t)B * n;
    gather_rows<<<(unsigned)((tot + 255) / 256), 256, 0, c.st>>>(srcT, c.fpsIdx, tOut, 1, B, Nsrc, n);
  }
  {
    int waves = B * n;
    knn_kernel<<<(unsigned)((waves * 32 + 127) / 128), 128, 0, c.st>>>(xOut, srcXyz, B, n, Nsrc, c.nind);
  }
  {
    size_t cols = (size_t)B * n * K_NN;
    group_kernel<<<(unsigned)((cols + 255) / 256), 256, 0, c.st>>>(srcXyz, srcT, srcF, xOut,
                                                                   c.nind, c.bufA, B, Nsrc, n, Cf);
  }
  float* nf = run_chain(c, conv, convC, 3, B * n * K_NN);  // ends in bufB
  int Cc = convC[3];
  float* u  = (nf == c.bufB) ? c.bufA : c.bufB;
  float* uy = (u == c.bufA) ? c.bufB : c.bufA;
  {
    size_t tot = (size_t)Cc * B * n;
    scatter_max<<<(unsigned)((tot + 255) / 256), 256, 0, c.st>>>(nf, srcT, c.nind, u, B, Nsrc, n, Cc);
  }
  launch_layer(c, unary, u, uy, 2 * Cc, cuOut, B * n, true, 0.01f);
  hipMemcpyAsync(pOut, uy, (size_t)cuOut * B * n * sizeof(float),
                 hipMemcpyDeviceToDevice, c.st);
}

static float* run_fp(Ctx& c, const float* xyz2, const float* xyz1,
                     const float* f2, const float* f1, int C2, int C1,
                     int N2, int N1, const ConvLayer fp[2], const int fpC[3],
                     float* pOut /*nullable*/) {
  int B = c.B;
  size_t cols = (size_t)B * N2;
  fp_interp<<<(unsigned)((cols + 127) / 128), 128, 0, c.st>>>(xyz2, xyz1, f1, f2, c.bufA,
                                                              B, N2, N1, C1, C2);
  float* r = run_chain(c, fp, fpC, 2, B * N2);  // ends back in bufA
  if (pOut)
    hipMemcpyAsync(pOut, r, (size_t)fpC[2] * cols * sizeof(float),
                   hipMemcpyDeviceToDevice, c.st);
  return r;
}

extern "C" void kernel_launch(void* const* d_in, const int* in_sizes, int n_in,
                              void* d_out, int out_size, void* d_ws, size_t ws_size,
                              hipStream_t stream) {
  (void)in_sizes; (void)n_in; (void)out_size; (void)ws_size;
  const int B = 2, N = 16384;

  // ---- inputs (setup_inputs dict insertion order; params leaves in order) ----
  int pi = 0;
  const float* xyz   = (const float*)d_in[pi++];
  const float* feat  = (const float*)d_in[pi++];
  const float* times = (const float*)d_in[pi++];
  auto nextf = [&]() { return (const float*)d_in[pi++]; };
  auto readL = [&](ConvLayer& L) { L.w = nextf(); L.b = nextf(); L.g = nextf(); L.bt = nextf(); };
  ConvLayer m1c[3], m2c[3], m3c[3], m4c[3];
  ConvLayer m1u, m2u, m3u, m4u;
  ConvLayer fp1[2], fp2[2], fp3[2], fp4[2];
  for (int l = 0; l < 3; ++l) readL(m1c[l]);  readL(m1u);
  for (int l = 0; l < 3; ++l) readL(m2c[l]);  readL(m2u);
  for (int l = 0; l < 3; ++l) readL(m3c[l]);  readL(m3u);
  for (int l = 0; l < 3; ++l) readL(m4c[l]);  readL(m4u);
  for (int l = 0; l < 2; ++l) readL(fp1[l]);
  for (int l = 0; l < 2; ++l) readL(fp2[l]);
  for (int l = 0; l < 2; ++l) readL(fp3[l]);
  for (int l = 0; l < 2; ++l) readL(fp4[l]);
  ConvLayer cls; cls.w = nextf(); cls.b = nextf(); cls.g = nullptr; cls.bt = nullptr;

  // ---- workspace bump allocator ----
  char* wp = (char*)d_ws;
  auto alloc = [&](size_t bytes) {
    void* p = (void*)wp;
    wp += (bytes + 255) & ~(size_t)255;
    return p;
  };
  size_t BN = (size_t)B * N;
  float* coords0 = (float*)alloc(3 * BN * 4);
  float* times0  = (float*)alloc(BN * 4);
  float* l0p     = (float*)alloc(2 * BN * 4);
  float *x1 = (float*)alloc(3 * (size_t)B * 2048 * 4), *t1 = (float*)alloc((size_t)B * 2048 * 4);
  float *x2 = (float*)alloc(3 * (size_t)B * 512 * 4),  *t2 = (float*)alloc((size_t)B * 512 * 4);
  float *x3 = (float*)alloc(3 * (size_t)B * 128 * 4),  *t3 = (float*)alloc((size_t)B * 128 * 4);
  float *x4 = (float*)alloc(3 * (size_t)B * 64 * 4),   *t4 = (float*)alloc((size_t)B * 64 * 4);
  float* p1 = (float*)alloc((size_t)128  * B * 2048 * 4);
  float* p2 = (float*)alloc((size_t)256  * B * 512 * 4);
  float* p3 = (float*)alloc((size_t)512  * B * 128 * 4);
  float* p4 = (float*)alloc((size_t)1024 * B * 64 * 4);

  Ctx c;
  c.st = stream; c.B = B;
  c.bufA   = (float*)alloc((size_t)128 * 131072 * 4);   // 64 MiB ping
  c.bufB   = (float*)alloc((size_t)128 * 131072 * 4);   // 64 MiB pong
  c.XhT    = (_Float16*)alloc((size_t)8 * 1024 * 1024 * 2);
  c.Wh     = (_Float16*)alloc((size_t)1024 * 2048 * 2);
  c.mean   = (float*)alloc(1024 * 4);
  c.istd   = (float*)alloc(1024 * 4);
  c.fpsDist= (float*)alloc(BN * 4);
  c.fpsIdx = (int*)alloc((size_t)B * 2048 * 4);
  c.nind   = (int*)alloc((size_t)B * 2048 * K_NN * 4);

  // ---- ingest ----
  ingest_kernel<<<(unsigned)((BN + 255) / 256), 256, 0, stream>>>(xyz, feat, times,
                                                                  coords0, times0, l0p, B, N);

  // ---- meteor stages ----
  static const int c1[4] = {6, 32, 32, 128};
  static const int c2c[4] = {132, 64, 64, 256};
  static const int c3c[4] = {260, 128, 128, 512};
  static const int c4c[4] = {516, 256, 256, 1024};
  run_meteor(c, coords0, times0, l0p, 2,   N,    2048, m1c, c1,  m1u, 128,  x1, t1, p1);
  run_meteor(c, x1,      t1,     p1, 128,  2048, 512,  m2c, c2c, m2u, 256,  x2, t2, p2);
  run_meteor(c, x2,      t2,     p2, 256,  512,  128,  m3c, c3c, m3u, 512,  x3, t3, p3);
  run_meteor(c, x3,      t3,     p3, 512,  128,  64,   m4c, c4c, m4u, 1024, x4, t4, p4);

  // ---- feature propagation ----
  static const int f1c[3] = {1536, 256, 256};
  static const int f2c[3] = {512, 256, 256};
  static const int f3c[3] = {384, 256, 128};
  static const int f4c[3] = {130, 128, 128};
  run_fp(c, x3, x4, p3, p4, 512, 1024, 128, 64, fp1, f1c, p3);   // p3 -> 256 ch
  run_fp(c, x2, x3, p2, p3, 256, 256, 512, 128, fp2, f2c, p2);   // p2 -> 256 ch
  run_fp(c, x1, x2, p1, p2, 128, 256, 2048, 512, fp3, f3c, p1);  // p1 -> 128 ch
  float* nf = run_fp(c, coords0, x1, l0p, p1, 2, 128, N, 2048, fp4, f4c, nullptr); // bufA

  // ---- classifier + slice ----
  float* ypred = (nf == c.bufA) ? c.bufB : c.bufA;
  launch_layer(c, cls, nf, ypred, 128, 20, B * N, false, 0.0f);
  {
    size_t tot = (size_t)B * 20 * (N / 2);
    slice_out<<<(unsigned)((tot + 255) / 256), 256, 0, stream>>>(ypred, (float*)d_out, B, N, 20);
  }
}
